// EGNNMixed2DEncoder_56581899157766
// MI455X (gfx1250) — matrix operand order repaired and checked by reading, not compiled, not run on hardware
//
#include <hip/hip_runtime.h>

// EGNN mixed 2D encoder for MI455X (gfx1250, wave32, WMMA).
// - em2 MLP is dead code (LayerNorm over a width-1 feature collapses to beta),
//   so the coordinate update is a cheap scatter.
// - All MLPs: f16-in / f32-acc v_wmma_f32_16x16x32_f16, weights resident in LDS
//   (pre-converted to f16 column-major once per launch).
// - Edge kernels: persistent workgroups, DOUBLE-BUFFERED A tiles filled with
//   GLOBAL_LOAD_ASYNC_TO_LDS_B128 (ASYNCcnt) overlapping the WMMA GEMMs.
// - h / h1 / h2 / edge_attr kept as f16 mirrors to halve gather traffic.

#define NNODE 20000
#define DDIM  128
#define EGN   500000
#define ELN   250000
#define NLAYER 5

typedef _Float16 half_t;
typedef __attribute__((ext_vector_type(16))) _Float16 v16h;
typedef __attribute__((ext_vector_type(8)))  _Float16 v8h;
typedef __attribute__((ext_vector_type(8)))  float    v8f;

__device__ __forceinline__ float siluf(float x) {
  return x * (1.0f / (1.0f + __expf(-x)));
}

__device__ __forceinline__ v8f wmma16(v16h a, v16h b, v8f c) {
  return __builtin_amdgcn_wmma_f32_16x16x32_f16(false, a, false, b, (short)0, c,
                                                false, false);
}

// Low 32 bits of a generic LDS pointer == LDS byte offset (ISA flat->LDS map).
__device__ __forceinline__ unsigned lds_off(const void* p) {
  return (unsigned)(unsigned long long)p;
}

// 16B global -> LDS async copy, tracked by ASYNCcnt.
__device__ __forceinline__ void async_cp16(unsigned ldsbyte, const void* gptr) {
  asm volatile("global_load_async_to_lds_b128 %0, %1, off"
               :
               : "v"(ldsbyte), "v"(gptr)
               : "memory");
}

__device__ __forceinline__ void wait_async0() {
  asm volatile("s_wait_asynccnt 0" ::: "memory");
}

// Contiguous f16 global -> LDS staging via async copies (halves % 8 == 0).
__device__ __forceinline__ void stage_lds_async(const half_t* __restrict__ g,
                                                half_t* lds, int halves,
                                                int tid) {
  unsigned o = lds_off(lds);
  for (int i = tid; i < (halves >> 3); i += 128)
    async_cp16(o + (unsigned)i * 16u, g + (size_t)i * 8);
}

// A fragment: rows r0..r0+15, 32-wide K chunk kc, row-major LDS tile.
__device__ __forceinline__ v16h ld_a_frag(const half_t* A, int ldk, int r0,
                                          int kc, int lane) {
  const half_t* p = A + (r0 + (lane & 15)) * ldk + kc * 32 + (lane >> 4) * 8;
  v8h lo = *(const v8h*)(p);
  v8h hi = *(const v8h*)(p + 16);
  return __builtin_shufflevector(lo, hi, 0, 1, 2, 3, 4, 5, 6, 7, 8, 9, 10, 11,
                                 12, 13, 14, 15);
}

// B fragment: cols n0..n0+15 from column-major LDS weights W[n*ldk + k].
__device__ __forceinline__ v16h ld_b_frag(const half_t* W, int ldk, int n0,
                                          int kc, int lane) {
  const half_t* p = W + (n0 + (lane & 15)) * ldk + kc * 32 + (lane >> 4) * 8;
  v8h lo = *(const v8h*)(p);
  v8h hi = *(const v8h*)(p + 16);
  return __builtin_shufflevector(lo, hi, 0, 1, 2, 3, 4, 5, 6, 7, 8, 9, 10, 11,
                                 12, 13, 14, 15);
}

// ---------------------------------------------------------------------------
// One-time prep kernels.
// ---------------------------------------------------------------------------
// fp32 row-major [K][N] -> f16 column-major [N][Kp] (zero pad K..Kp).
__global__ void k_wt(const float* __restrict__ src, half_t* __restrict__ dst,
                     int K, int N, int Kp) {
  int idx = blockIdx.x * blockDim.x + threadIdx.x;
  if (idx < N * Kp) {
    int n = idx / Kp, k = idx - n * Kp;
    dst[idx] = (half_t)((k < K) ? src[k * N + n] : 0.0f);
  }
}

__global__ void k_f32_to_f16(const float* __restrict__ src,
                             half_t* __restrict__ dst, int n) {
  int i = blockIdx.x * blockDim.x + threadIdx.x;
  if (i < n) dst[i] = (half_t)src[i];
}

// eal[e][0:128]=ear[e], eal[e][128:256]=eap[e] (f16).
__global__ void k_build_eal(const float* __restrict__ ear,
                            const float* __restrict__ eap,
                            half_t* __restrict__ eal) {
  int i = blockIdx.x * blockDim.x + threadIdx.x;
  if (i < ELN * 256) {
    int e = i >> 8, k = i & 255;
    float v = (k < 128) ? ear[e * 128 + k] : eap[e * 128 + (k - 128)];
    eal[i] = (half_t)v;
  }
}

__global__ void k_deg(const int* __restrict__ eg, float* __restrict__ cnt) {
  int e = blockIdx.x * blockDim.x + threadIdx.x;
  if (e < EGN) atomicAdd(&cnt[eg[EGN + e]], 1.0f);
}

__global__ void k_possrc(const int* __restrict__ eg,
                         const float* __restrict__ pos,
                         float* __restrict__ psum) {
  int e = blockIdx.x * blockDim.x + threadIdx.x;
  if (e < EGN) {
    int s = eg[e], d = eg[EGN + e];
    atomicAdd(&psum[d * 3 + 0], pos[s * 3 + 0]);
    atomicAdd(&psum[d * 3 + 1], pos[s * 3 + 1]);
    atomicAdd(&psum[d * 3 + 2], pos[s * 3 + 2]);
  }
}

// ---------------------------------------------------------------------------
// EGNN edge MLP: feat[EG,257] -> 256 (SiLU) -> 64 (SiLU) -> LN(64) -> atomic
// scatter into m1_sum[dst]. Double-buffered async A-tile gather.
// ---------------------------------------------------------------------------
__device__ __forceinline__ void egnn_prefetch(int e0, const int* __restrict__ eg,
                                              const half_t* __restrict__ hh,
                                              const float* __restrict__ pos,
                                              half_t* A, int* dI, int* sI,
                                              int tid) {
  unsigned ao = lds_off(A);
  // 64 rows x 32 chunks of 16B: c<16 -> hh[dst] (k 0..127), c>=16 -> hh[src].
  for (int idx = tid; idx < 64 * 32; idx += 128) {
    int r = idx >> 5, c = idx & 31;
    int e = e0 + r;
    int nidx = 0;
    if (e < EGN) nidx = (c < 16) ? eg[EGN + e] : eg[e];
    unsigned dstoff =
        ao + (unsigned)(r * 288 + ((c & 15) << 3) + ((c >= 16) ? 128 : 0)) * 2u;
    async_cp16(dstoff, hh + (size_t)nidx * DDIM + ((c & 15) << 3));
  }
  if (tid < 64) {
    int e = e0 + tid;
    if (e < EGN) {
      int s = eg[e], d = eg[EGN + e];
      float dx = pos[s * 3 + 0] - pos[d * 3 + 0];
      float dy = pos[s * 3 + 1] - pos[d * 3 + 1];
      float dz = pos[s * 3 + 2] - pos[d * 3 + 2];
      sI[tid] = s; dI[tid] = d;
      A[tid * 288 + 256] = (half_t)(dx * dx + dy * dy + dz * dz);
    } else {
      sI[tid] = 0; dI[tid] = -1;
      A[tid * 288 + 256] = (half_t)0.0f;
    }
  }
}

__global__ __launch_bounds__(128, 1) void k_egnn_edge(
    const half_t* __restrict__ hh, const float* __restrict__ pos,
    const int* __restrict__ eg, const half_t* __restrict__ w1t,
    const float* __restrict__ b1, const half_t* __restrict__ w2t,
    const float* __restrict__ b2, const float* __restrict__ lng,
    const float* __restrict__ lnb, float* __restrict__ m1_sum) {
  constexpr int K1P = 288, N1 = 256, N2 = 64, K2 = 256;
  extern __shared__ char smem[];
  half_t* W1 = (half_t*)smem;          // 256*288
  half_t* W2 = W1 + N1 * K1P;          // 64*256
  half_t* A0 = W2 + N2 * K2;           // 64*288
  half_t* A1 = A0 + 64 * K1P;          // 64*288
  half_t* H  = A1 + 64 * K1P;          // 64*256
  float*  Mf = (float*)(H + 64 * N1);  // 64*64
  float*  B1 = Mf + 64 * 64;
  float*  B2 = B1 + N1;
  float*  G1 = B2 + N2;
  float*  Bb = G1 + N2;
  int*    dI0 = (int*)(Bb + N2);
  int*    sI0 = dI0 + 64;
  int*    dI1 = sI0 + 64;
  int*    sI1 = dI1 + 64;

  const int tid = threadIdx.x;
  stage_lds_async(w1t, W1, N1 * K1P, tid);
  stage_lds_async(w2t, W2, N2 * K2, tid);
  for (int i = tid; i < N1; i += 128) B1[i] = b1[i];
  for (int i = tid; i < N2; i += 128) {
    B2[i] = b2[i]; G1[i] = lng[i]; Bb[i] = lnb[i];
  }
  // zero the pad columns (257..287) once; col 256 (dist) is written per tile.
  for (int i = tid; i < 64 * 32; i += 128) {
    int r = i >> 5, k = 256 + (i & 31);
    A0[r * K1P + k] = (half_t)0.0f;
    A1[r * K1P + k] = (half_t)0.0f;
  }

  const int wv = tid >> 5, lane = tid & 31;
  const int r0 = wv * 16;
  const int ntiles = (EGN + 63) / 64;
  const int stride = gridDim.x;

  int buf = 0;
  int t = blockIdx.x;
  if (t < ntiles) egnn_prefetch(t * 64, eg, hh, pos, A0, dI0, sI0, tid);

  for (; t < ntiles; t += stride) {
    wait_async0();
    __syncthreads();
    half_t* A = buf ? A1 : A0;
    int* dI = buf ? dI1 : dI0;
    int tn = t + stride;
    if (tn < ntiles)
      egnn_prefetch(tn * 64, eg, hh, pos, buf ? A0 : A1, buf ? dI0 : dI1,
                    buf ? sI0 : sI1, tid);
    // GEMM1: [64x288] @ [288x256] -> SiLU -> H
    for (int nt = 0; nt < 16; ++nt) {
      v8f acc = {};
#pragma unroll
      for (int kc = 0; kc < 9; ++kc)
        acc = wmma16(ld_a_frag(A, K1P, r0, kc, lane),
                     ld_b_frag(W1, K1P, nt * 16, kc, lane), acc);
      int n = nt * 16 + (lane & 15);
      float bias = B1[n];
      int mb = r0 + ((lane >> 4) << 3);
#pragma unroll
      for (int r = 0; r < 8; ++r)
        H[(mb + r) * N1 + n] = (half_t)siluf(acc[r] + bias);
    }
    __syncthreads();
    // GEMM2: [64x256] @ [256x64] -> SiLU -> Mf (f32)
    for (int nt = 0; nt < 4; ++nt) {
      v8f acc = {};
#pragma unroll
      for (int kc = 0; kc < 8; ++kc)
        acc = wmma16(ld_a_frag(H, N1, r0, kc, lane),
                     ld_b_frag(W2, K2, nt * 16, kc, lane), acc);
      int n = nt * 16 + (lane & 15);
      float bias = B2[n];
      int mb = r0 + ((lane >> 4) << 3);
#pragma unroll
      for (int r = 0; r < 8; ++r)
        Mf[(mb + r) * 64 + n] = siluf(acc[r] + bias);
    }
    __syncthreads();
    // LayerNorm(64) per edge + atomic scatter-sum into m1_sum[dst].
    if (tid < 64) {
      int d = dI[tid];
      if (d >= 0) {
        const float* row = Mf + tid * 64;
        float mu = 0.0f;
        for (int k = 0; k < 64; ++k) mu += row[k];
        mu *= (1.0f / 64.0f);
        float var = 0.0f;
        for (int k = 0; k < 64; ++k) { float c = row[k] - mu; var += c * c; }
        float rs = rsqrtf(var * (1.0f / 64.0f) + 1e-5f);
        for (int k = 0; k < 64; ++k)
          atomicAdd(&m1_sum[d * 64 + k], (row[k] - mu) * rs * G1[k] + Bb[k]);
      }
    }
    buf ^= 1;
  }
}

// ---------------------------------------------------------------------------
// GINE edge MLP: eal (256) -> 256 (SiLU) -> 128; msg = silu(h_src*eattr);
// atomic scatter into agg[dst]. Double-buffered async A-tile copy.
// ---------------------------------------------------------------------------
__device__ __forceinline__ void gine_prefetch(int e0, const int* __restrict__ el,
                                              const half_t* __restrict__ eal,
                                              half_t* A, int* dI, int* sI,
                                              int tid) {
  unsigned ao = lds_off(A);
  for (int idx = tid; idx < 64 * 32; idx += 128) {
    int r = idx >> 5, c = idx & 31;
    int e = e0 + r;
    if (e >= ELN) e = 0;
    async_cp16(ao + (unsigned)(r * 256 + (c << 3)) * 2u,
               eal + ((size_t)e << 8) + (c << 3));
  }
  if (tid < 64) {
    int e = e0 + tid;
    if (e < ELN) { sI[tid] = el[e]; dI[tid] = el[ELN + e]; }
    else         { sI[tid] = 0;     dI[tid] = -1; }
  }
}

__global__ __launch_bounds__(128, 1) void k_gine_edge(
    const half_t* __restrict__ hh, const half_t* __restrict__ eal,
    const int* __restrict__ el, const half_t* __restrict__ w1t,
    const float* __restrict__ b1, const half_t* __restrict__ w2t,
    const float* __restrict__ b2, float* __restrict__ agg) {
  constexpr int K1 = 256, N1 = 256, K2 = 256, N2 = 128;
  extern __shared__ char smem[];
  half_t* W1 = (half_t*)smem;
  half_t* W2 = W1 + N1 * K1;
  half_t* A0 = W2 + N2 * K2;
  half_t* A1 = A0 + 64 * K1;
  half_t* H  = A1 + 64 * K1;
  float*  B1 = (float*)(H + 64 * N1);
  float*  B2 = B1 + N1;
  int*    dI0 = (int*)(B2 + N2);
  int*    sI0 = dI0 + 64;
  int*    dI1 = sI0 + 64;
  int*    sI1 = dI1 + 64;

  const int tid = threadIdx.x;
  stage_lds_async(w1t, W1, N1 * K1, tid);
  stage_lds_async(w2t, W2, N2 * K2, tid);
  for (int i = tid; i < N1; i += 128) B1[i] = b1[i];
  for (int i = tid; i < N2; i += 128) B2[i] = b2[i];

  const int wv = tid >> 5, lane = tid & 31;
  const int r0 = wv * 16;
  const int ntiles = (ELN + 63) / 64;
  const int stride = gridDim.x;

  int buf = 0;
  int t = blockIdx.x;
  if (t < ntiles) gine_prefetch(t * 64, el, eal, A0, dI0, sI0, tid);

  for (; t < ntiles; t += stride) {
    wait_async0();
    __syncthreads();
    half_t* A = buf ? A1 : A0;
    int* dI = buf ? dI1 : dI0;
    int* sI = buf ? sI1 : sI0;
    int tn = t + stride;
    if (tn < ntiles)
      gine_prefetch(tn * 64, el, eal, buf ? A0 : A1, buf ? dI0 : dI1,
                    buf ? sI0 : sI1, tid);
    for (int nt = 0; nt < 16; ++nt) {
      v8f acc = {};
#pragma unroll
      for (int kc = 0; kc < 8; ++kc)
        acc = wmma16(ld_a_frag(A, K1, r0, kc, lane),
                     ld_b_frag(W1, K1, nt * 16, kc, lane), acc);
      int n = nt * 16 + (lane & 15);
      float bias = B1[n];
      int mb = r0 + ((lane >> 4) << 3);
#pragma unroll
      for (int r = 0; r < 8; ++r)
        H[(mb + r) * N1 + n] = (half_t)siluf(acc[r] + bias);
    }
    __syncthreads();
    for (int nt = 0; nt < 8; ++nt) {
      v8f acc = {};
#pragma unroll
      for (int kc = 0; kc < 8; ++kc)
        acc = wmma16(ld_a_frag(H, N1, r0, kc, lane),
                     ld_b_frag(W2, K2, nt * 16, kc, lane), acc);
      int n = nt * 16 + (lane & 15);
      float bias = B2[n];
      int mb = r0 + ((lane >> 4) << 3);
#pragma unroll
      for (int r = 0; r < 8; ++r) {
        int m = mb + r, d = dI[m];
        if (d >= 0) {
          float eattr = acc[r] + bias;
          float hv = (float)hh[(size_t)sI[m] * DDIM + n];
          atomicAdd(&agg[d * DDIM + n], siluf(hv * eattr));
        }
      }
    }
    __syncthreads();
    buf ^= 1;
  }
}

// ---------------------------------------------------------------------------
// Node update: upd = LN(nm([LN(h) | m1_mean])); h1h = f16(h + upd); pos update.
// ---------------------------------------------------------------------------
__global__ __launch_bounds__(128, 1) void k_node(
    const float* __restrict__ h, const float* __restrict__ m1s,
    const float* __restrict__ cnt, const float* __restrict__ psum,
    float* __restrict__ pos, const half_t* __restrict__ w1t,
    const float* __restrict__ b1, const half_t* __restrict__ w2t,
    const float* __restrict__ b2, const float* __restrict__ g1,
    const float* __restrict__ bb1, const float* __restrict__ g2,
    const float* __restrict__ bb2, const float* __restrict__ en2b,
    half_t* __restrict__ h1h) {
  constexpr int K1 = 192, N1 = 256, K2 = 256, N2 = 128;
  extern __shared__ char smem[];
  half_t* W1 = (half_t*)smem;
  half_t* W2 = W1 + N1 * K1;
  half_t* A  = W2 + N2 * K2;
  half_t* H  = A + 64 * K1;
  float*  Uf  = (float*)(H + 64 * N1);
  float*  B1  = Uf + 64 * N2;
  float*  B2  = B1 + N1;
  float*  G1  = B2 + N2;
  float*  Bb1 = G1 + 128;
  float*  G2  = Bb1 + 128;
  float*  Bb2 = G2 + 128;

  const int tid = threadIdx.x;
  stage_lds_async(w1t, W1, N1 * K1, tid);
  stage_lds_async(w2t, W2, N2 * K2, tid);
  for (int i = tid; i < N1; i += 128) B1[i] = b1[i];
  for (int i = tid; i < N2; i += 128) {
    B2[i] = b2[i]; G1[i] = g1[i]; Bb1[i] = bb1[i]; G2[i] = g2[i]; Bb2[i] = bb2[i];
  }
  wait_async0();
  __syncthreads();

  const int n0 = blockIdx.x * 64;
  if (tid < 64) {
    int r = n0 + tid;
    half_t* ar = A + tid * K1;
    if (r < NNODE) {
      const float* hr = h + r * DDIM;
      float mu = 0.0f;
      for (int k = 0; k < 128; ++k) mu += hr[k];
      mu *= (1.0f / 128.0f);
      float var = 0.0f;
      for (int k = 0; k < 128; ++k) { float c = hr[k] - mu; var += c * c; }
      float rs = rsqrtf(var * (1.0f / 128.0f) + 1e-5f);
      for (int k = 0; k < 128; ++k)
        ar[k] = (half_t)((hr[k] - mu) * rs * G1[k] + Bb1[k]);
      float inv = 1.0f / fmaxf(cnt[r], 1.0f);
      for (int k = 0; k < 64; ++k) ar[128 + k] = (half_t)(m1s[r * 64 + k] * inv);
    } else {
      for (int k = 0; k < K1; ++k) ar[k] = (half_t)0.0f;
    }
  }
  __syncthreads();

  const int wv = tid >> 5, lane = tid & 31;
  const int r0 = wv * 16;
  for (int nt = 0; nt < 16; ++nt) {
    v8f acc = {};
#pragma unroll
    for (int kc = 0; kc < 6; ++kc)
      acc = wmma16(ld_a_frag(A, K1, r0, kc, lane),
                   ld_b_frag(W1, K1, nt * 16, kc, lane), acc);
    int n = nt * 16 + (lane & 15);
    float bias = B1[n];
    int mb = r0 + ((lane >> 4) << 3);
#pragma unroll
    for (int r = 0; r < 8; ++r)
      H[(mb + r) * N1 + n] = (half_t)siluf(acc[r] + bias);
  }
  __syncthreads();
  for (int nt = 0; nt < 8; ++nt) {
    v8f acc = {};
#pragma unroll
    for (int kc = 0; kc < 8; ++kc)
      acc = wmma16(ld_a_frag(H, N1, r0, kc, lane),
                   ld_b_frag(W2, K2, nt * 16, kc, lane), acc);
    int n = nt * 16 + (lane & 15);
    float bias = B2[n];
    int mb = r0 + ((lane >> 4) << 3);
#pragma unroll
    for (int r = 0; r < 8; ++r) Uf[(mb + r) * N2 + n] = acc[r] + bias;
  }
  __syncthreads();
  if (tid < 64) {
    int r = n0 + tid;
    if (r < NNODE) {
      const float* u = Uf + tid * 128;
      float mu = 0.0f;
      for (int k = 0; k < 128; ++k) mu += u[k];
      mu *= (1.0f / 128.0f);
      float var = 0.0f;
      for (int k = 0; k < 128; ++k) { float c = u[k] - mu; var += c * c; }
      float rs = rsqrtf(var * (1.0f / 128.0f) + 1e-5f);
      for (int k = 0; k < 128; ++k)
        h1h[(size_t)r * DDIM + k] =
            (half_t)(h[r * DDIM + k] + (u[k] - mu) * rs * G2[k] + Bb2[k]);
      // pos update: m2_i = en2_b * (pos_i - mean_src(pos)); zero if deg==0.
      float c = cnt[r];
      if (c > 0.0f) {
        float m2b = en2b[0];
        float inv = 1.0f / c;
        for (int d3 = 0; d3 < 3; ++d3) {
          float p = pos[r * 3 + d3];
          pos[r * 3 + d3] = p + m2b * (p - psum[r * 3 + d3] * inv);
        }
      }
    }
  }
}

// ---------------------------------------------------------------------------
// GINE fin layer: h2h = f16(fl(agg + h)), 128 -> 256 (SiLU) -> 128.
// ---------------------------------------------------------------------------
__global__ __launch_bounds__(128, 1) void k_fl(
    const float* __restrict__ h, const float* __restrict__ agg,
    const half_t* __restrict__ w1t, const float* __restrict__ b1,
    const half_t* __restrict__ w2t, const float* __restrict__ b2,
    half_t* __restrict__ h2h) {
  constexpr int K1 = 128, N1 = 256, K2 = 256, N2 = 128;
  extern __shared__ char smem[];
  half_t* W1 = (half_t*)smem;
  half_t* W2 = W1 + N1 * K1;
  half_t* A  = W2 + N2 * K2;
  half_t* H  = A + 64 * K1;
  float*  B1 = (float*)(H + 64 * N1);
  float*  B2 = B1 + N1;

  const int tid = threadIdx.x;
  stage_lds_async(w1t, W1, N1 * K1, tid);
  stage_lds_async(w2t, W2, N2 * K2, tid);
  for (int i = tid; i < N1; i += 128) B1[i] = b1[i];
  for (int i = tid; i < N2; i += 128) B2[i] = b2[i];

  const int n0 = blockIdx.x * 64;
  for (int idx = tid; idx < 64 * 128; idx += 128) {
    int r = idx >> 7, k = idx & 127, gr = n0 + r;
    float v = (gr < NNODE) ? (agg[gr * DDIM + k] + h[gr * DDIM + k]) : 0.0f;
    A[idx] = (half_t)v;
  }
  wait_async0();
  __syncthreads();

  const int wv = tid >> 5, lane = tid & 31;
  const int r0 = wv * 16;
  for (int nt = 0; nt < 16; ++nt) {
    v8f acc = {};
#pragma unroll
    for (int kc = 0; kc < 4; ++kc)
      acc = wmma16(ld_a_frag(A, K1, r0, kc, lane),
                   ld_b_frag(W1, K1, nt * 16, kc, lane), acc);
    int n = nt * 16 + (lane & 15);
    float bias = B1[n];
    int mb = r0 + ((lane >> 4) << 3);
#pragma unroll
    for (int r = 0; r < 8; ++r)
      H[(mb + r) * N1 + n] = (half_t)siluf(acc[r] + bias);
  }
  __syncthreads();
  for (int nt = 0; nt < 8; ++nt) {
    v8f acc = {};
#pragma unroll
    for (int kc = 0; kc < 8; ++kc)
      acc = wmma16(ld_a_frag(H, N1, r0, kc, lane),
                   ld_b_frag(W2, K2, nt * 16, kc, lane), acc);
    int n = nt * 16 + (lane & 15);
    float bias = B2[n];
    int mb = r0 + ((lane >> 4) << 3);
#pragma unroll
    for (int r = 0; r < 8; ++r) {
      int gr = n0 + mb + r;
      if (gr < NNODE) h2h[(size_t)gr * DDIM + n] = (half_t)(acc[r] + bias);
    }
  }
}

// ---------------------------------------------------------------------------
// Combine: h += lc([h1h | h2h]); also refresh hh (f16 mirror of h).
// ---------------------------------------------------------------------------
__global__ __launch_bounds__(128, 1) void k_lc(
    float* __restrict__ h, half_t* __restrict__ hh,
    const half_t* __restrict__ h1h, const half_t* __restrict__ h2h,
    const half_t* __restrict__ w1t, const float* __restrict__ b1,
    const half_t* __restrict__ w2t, const float* __restrict__ b2) {
  constexpr int K1 = 256, N1 = 256, K2 = 256, N2 = 128;
  extern __shared__ char smem[];
  half_t* W1 = (half_t*)smem;
  half_t* W2 = W1 + N1 * K1;
  half_t* A  = W2 + N2 * K2;
  half_t* H  = A + 64 * K1;
  float*  B1 = (float*)(H + 64 * N1);
  float*  B2 = B1 + N1;

  const int tid = threadIdx.x;
  stage_lds_async(w1t, W1, N1 * K1, tid);
  stage_lds_async(w2t, W2, N2 * K2, tid);
  for (int i = tid; i < N1; i += 128) B1[i] = b1[i];
  for (int i = tid; i < N2; i += 128) B2[i] = b2[i];

  const int n0 = blockIdx.x * 64;
  {
    unsigned ao = lds_off(A);
    for (int idx = tid; idx < 64 * 32; idx += 128) {
      int r = idx >> 5, c = idx & 31;
      int gr = n0 + r;
      if (gr >= NNODE) gr = 0;
      const half_t* g = (c < 16) ? (h1h + (size_t)gr * DDIM + ((c & 15) << 3))
                                 : (h2h + (size_t)gr * DDIM + ((c & 15) << 3));
      unsigned doff =
          ao + (unsigned)(r * 256 + ((c & 15) << 3) + ((c >= 16) ? 128 : 0)) * 2u;
      async_cp16(doff, g);
    }
  }
  wait_async0();
  __syncthreads();

  const int wv = tid >> 5, lane = tid & 31;
  const int r0 = wv * 16;
  for (int nt = 0; nt < 16; ++nt) {
    v8f acc = {};
#pragma unroll
    for (int kc = 0; kc < 8; ++kc)
      acc = wmma16(ld_a_frag(A, K1, r0, kc, lane),
                   ld_b_frag(W1, K1, nt * 16, kc, lane), acc);
    int n = nt * 16 + (lane & 15);
    float bias = B1[n];
    int mb = r0 + ((lane >> 4) << 3);
#pragma unroll
    for (int r = 0; r < 8; ++r)
      H[(mb + r) * N1 + n] = (half_t)siluf(acc[r] + bias);
  }
  __syncthreads();
  for (int nt = 0; nt < 8; ++nt) {
    v8f acc = {};
#pragma unroll
    for (int kc = 0; kc < 8; ++kc)
      acc = wmma16(ld_a_frag(H, N1, r0, kc, lane),
                   ld_b_frag(W2, K2, nt * 16, kc, lane), acc);
    int n = nt * 16 + (lane & 15);
    float bias = B2[n];
    int mb = r0 + ((lane >> 4) << 3);
#pragma unroll
    for (int r = 0; r < 8; ++r) {
      int gr = n0 + mb + r;
      if (gr < NNODE) {
        float nv = h[gr * DDIM + n] + acc[r] + bias;
        h[gr * DDIM + n] = nv;
        hh[(size_t)gr * DDIM + n] = (half_t)nv;
      }
    }
  }
}

// ---------------------------------------------------------------------------
// Host orchestration (graph-capture safe).
// ---------------------------------------------------------------------------
static constexpr size_t LDS_EDGE =
    (size_t)(256 * 288 + 64 * 256 + 2 * 64 * 288 + 64 * 256) * 2 +
    (size_t)(64 * 64 + 256 + 3 * 64 + 64) * 4 + (size_t)256 * 4;
static constexpr size_t LDS_GINE =
    (size_t)(256 * 256 + 128 * 256 + 2 * 64 * 256 + 64 * 256) * 2 +
    (size_t)(256 + 128) * 4 + (size_t)256 * 4;
static constexpr size_t LDS_NODE =
    (size_t)(256 * 192 + 128 * 256 + 64 * 192 + 64 * 256) * 2 +
    (size_t)(64 * 128 + 256 + 128 + 4 * 128) * 4;
static constexpr size_t LDS_FL =
    (size_t)(256 * 128 + 128 * 256 + 64 * 128 + 64 * 256) * 2 +
    (size_t)(256 + 128) * 4;
static constexpr size_t LDS_LC =
    (size_t)(256 * 256 + 128 * 256 + 64 * 256 + 64 * 256) * 2 +
    (size_t)(256 + 128) * 4;

extern "C" void kernel_launch(void* const* d_in, const int* in_sizes, int n_in,
                              void* d_out, int out_size, void* d_ws,
                              size_t ws_size, hipStream_t stream) {
  (void)in_sizes; (void)n_in; (void)out_size; (void)ws_size;
  const float* node = (const float*)d_in[0];
  const float* pos0 = (const float*)d_in[1];
  const int*   eg   = (const int*)d_in[2];
  const int*   el   = (const int*)d_in[3];
  const float* ear  = (const float*)d_in[4];
  const float* eap  = (const float*)d_in[5];

  // Params flattened in jax pytree (alphabetical dict key) order after the 6
  // positional inputs. em2_* and en2_g are dead (LN over width-1 feature).
  const float* P[32];
  for (int i = 0; i < 32; ++i) P[i] = (const float*)d_in[6 + i];
  const float *ec_b1 = P[0], *ec_b2 = P[1], *ec_w1 = P[2], *ec_w2 = P[3];
  const float *em1_b1 = P[4], *em1_b2 = P[5], *em1_w1 = P[6], *em1_w2 = P[7];
  const float *en1_b = P[12], *en1_g = P[13], *en2_b = P[14];
  const float *fl_b1 = P[16], *fl_b2 = P[17], *fl_w1 = P[18], *fl_w2 = P[19];
  const float *lc_b1 = P[20], *lc_b2 = P[21], *lc_w1 = P[22], *lc_w2 = P[23];
  const float *nm_b1 = P[24], *nm_b2 = P[25], *nm_w1 = P[26], *nm_w2 = P[27];
  const float *nn1_b = P[28], *nn1_g = P[29], *nn2_b = P[30], *nn2_g = P[31];

  // ---- workspace carve (256B aligned blocks) ----
  char* cur = (char*)d_ws;
  auto alloc_f = [&](size_t n) {
    float* p = (float*)cur;
    cur += ((n * 4 + 255) & ~(size_t)255);
    return p;
  };
  auto alloc_h = [&](size_t n) {
    half_t* p = (half_t*)cur;
    cur += ((n * 2 + 255) & ~(size_t)255);
    return p;
  };
  float* pos  = alloc_f((size_t)NNODE * 3);
  float* cnt  = alloc_f(NNODE);
  float* psum = alloc_f((size_t)NNODE * 3);
  float* m1s  = alloc_f((size_t)NNODE * 64);
  float* agg  = alloc_f((size_t)NNODE * 128);
  half_t* hh  = alloc_h((size_t)NNODE * 128);
  half_t* h1h = alloc_h((size_t)NNODE * 128);
  half_t* h2h = alloc_h((size_t)NNODE * 128);
  half_t* eal = alloc_h((size_t)ELN * 256);
  half_t* em1w1T = alloc_h((size_t)NLAYER * 256 * 288);
  half_t* em1w2T = alloc_h((size_t)NLAYER * 64 * 256);
  half_t* nmw1T  = alloc_h((size_t)NLAYER * 256 * 192);
  half_t* nmw2T  = alloc_h((size_t)NLAYER * 128 * 256);
  half_t* ecw1T  = alloc_h((size_t)NLAYER * 256 * 256);
  half_t* ecw2T  = alloc_h((size_t)NLAYER * 128 * 256);
  half_t* flw1T  = alloc_h((size_t)NLAYER * 256 * 128);
  half_t* flw2T  = alloc_h((size_t)NLAYER * 128 * 256);
  half_t* lcw1T  = alloc_h((size_t)256 * 256);
  half_t* lcw2T  = alloc_h((size_t)128 * 256);

  float* h = (float*)d_out;  // node features live in d_out throughout

  // ---- one-time prep ----
  hipMemcpyAsync(h, node, sizeof(float) * NNODE * DDIM,
                 hipMemcpyDeviceToDevice, stream);
  hipMemcpyAsync(pos, pos0, sizeof(float) * NNODE * 3,
                 hipMemcpyDeviceToDevice, stream);
  hipMemsetAsync(cnt, 0, sizeof(float) * NNODE, stream);
  k_f32_to_f16<<<dim3((NNODE * DDIM + 255) / 256), dim3(256), 0, stream>>>(
      node, hh, NNODE * DDIM);
  k_build_eal<<<dim3((ELN * 256 + 255) / 256), dim3(256), 0, stream>>>(ear, eap,
                                                                       eal);
  k_deg<<<dim3((EGN + 255) / 256), dim3(256), 0, stream>>>(eg, cnt);

  auto wt = [&](const float* src, half_t* dst, int K, int N, int Kp) {
    k_wt<<<dim3((N * Kp + 255) / 256), dim3(256), 0, stream>>>(src, dst, K, N,
                                                               Kp);
  };
  for (int l = 0; l < NLAYER; ++l) {
    wt(em1_w1 + (size_t)l * 257 * 256, em1w1T + (size_t)l * 256 * 288, 257, 256, 288);
    wt(em1_w2 + (size_t)l * 256 * 64,  em1w2T + (size_t)l * 64 * 256,  256, 64,  256);
    wt(nm_w1  + (size_t)l * 192 * 256, nmw1T  + (size_t)l * 256 * 192, 192, 256, 192);
    wt(nm_w2  + (size_t)l * 256 * 128, nmw2T  + (size_t)l * 128 * 256, 256, 128, 256);
    wt(ec_w1  + (size_t)l * 256 * 256, ecw1T  + (size_t)l * 256 * 256, 256, 256, 256);
    wt(ec_w2  + (size_t)l * 256 * 128, ecw2T  + (size_t)l * 128 * 256, 256, 128, 256);
    wt(fl_w1  + (size_t)l * 128 * 256, flw1T  + (size_t)l * 256 * 128, 128, 256, 128);
    wt(fl_w2  + (size_t)l * 256 * 128, flw2T  + (size_t)l * 128 * 256, 256, 128, 256);
  }
  wt(lc_w1, lcw1T, 256, 256, 256);
  wt(lc_w2, lcw2T, 256, 128, 256);

  const int nodeBlocks = (NNODE + 63) / 64;
  for (int l = 0; l < NLAYER; ++l) {
    hipMemsetAsync(m1s, 0, sizeof(float) * NNODE * 64, stream);
    hipMemsetAsync(psum, 0, sizeof(float) * NNODE * 3, stream);
    hipMemsetAsync(agg, 0, sizeof(float) * NNODE * 128, stream);

    k_egnn_edge<<<dim3(1024), dim3(128), LDS_EDGE, stream>>>(
        hh, pos, eg, em1w1T + (size_t)l * 256 * 288, em1_b1 + l * 256,
        em1w2T + (size_t)l * 64 * 256, em1_b2 + l * 64, en1_g + l * 64,
        en1_b + l * 64, m1s);
    k_possrc<<<dim3((EGN + 255) / 256), dim3(256), 0, stream>>>(eg, pos, psum);
    k_node<<<dim3(nodeBlocks), dim3(128), LDS_NODE, stream>>>(
        h, m1s, cnt, psum, pos, nmw1T + (size_t)l * 256 * 192, nm_b1 + l * 256,
        nmw2T + (size_t)l * 128 * 256, nm_b2 + l * 128, nn1_g + l * 128,
        nn1_b + l * 128, nn2_g + l * 128, nn2_b + l * 128, en2_b + l, h1h);
    k_gine_edge<<<dim3(1024), dim3(128), LDS_GINE, stream>>>(
        hh, eal, el, ecw1T + (size_t)l * 256 * 256, ec_b1 + l * 256,
        ecw2T + (size_t)l * 128 * 256, ec_b2 + l * 128, agg);
    k_fl<<<dim3(nodeBlocks), dim3(128), LDS_FL, stream>>>(
        h, agg, flw1T + (size_t)l * 256 * 128, fl_b1 + l * 256,
        flw2T + (size_t)l * 128 * 256, fl_b2 + l * 128, h2h);
    k_lc<<<dim3(nodeBlocks), dim3(128), LDS_LC, stream>>>(
        h, hh, h1h, h2h, lcw1T, lc_b1, lcw2T, lc_b2);
  }
}
